// FSAPhonotacticsModel_21234318312111
// MI455X (gfx1250) — matrix operand order repaired and checked by reading, not compiled, not run on hardware
//
#include <hip/hip_runtime.h>
#include <hip/hip_bf16.h>
#include <math.h>

// Problem constants (match reference)
#define QS 512   // FSA states
#define SS 64    // alphabet
#define BB 64    // batch
#define TT 256   // timesteps

typedef __attribute__((ext_vector_type(16))) _Float16 v16h;
typedef __attribute__((ext_vector_type(8)))  _Float16 v8h;
typedef __attribute__((ext_vector_type(8)))  float    v8f;

// ---------------------------------------------------------------------------
// Prep 1: tiled transpose+exp:  Mt[s][j][i] = (f16) exp(A[i][s][j])
// (32 MB f16, L2-resident; coalesced reads of A along j, coalesced writes
//  of Mt along i via a 32x32 LDS tile)
// ---------------------------------------------------------------------------
__global__ __launch_bounds__(256) void fsa_prep_A(const float* __restrict__ A,
                                                  _Float16* __restrict__ Mt) {
  __shared__ float tile[32][33];
  const int sIdx = blockIdx.z;          // symbol
  const int it = blockIdx.x * 32;       // i-tile base (output-state dim)
  const int jt = blockIdx.y * 32;       // j-tile base (source-state dim)
  const int tx = threadIdx.x & 31;
  const int ty = threadIdx.x >> 5;      // 0..7

  #pragma unroll
  for (int r = 0; r < 4; ++r) {
    const int i = it + ty + 8 * r;
    tile[ty + 8 * r][tx] = __expf(A[((size_t)i * SS + sIdx) * QS + jt + tx]);
  }
  __syncthreads();
  #pragma unroll
  for (int r = 0; r < 4; ++r) {
    const int j = jt + ty + 8 * r;
    Mt[((size_t)sIdx * QS + j) * QS + it + tx] = (_Float16)tile[tx][ty + 8 * r];
  }
}

// ---------------------------------------------------------------------------
// Prep 2: exp(init), exp(final)
// ---------------------------------------------------------------------------
__global__ __launch_bounds__(512) void fsa_prep_vec(const float* __restrict__ initv,
                                                    const float* __restrict__ finv,
                                                    float* __restrict__ initexp,
                                                    float* __restrict__ finexp) {
  const int i = threadIdx.x;
  if (i < QS) {
    initexp[i] = __expf(initv[i]);
    finexp[i]  = __expf(finv[i]);
  }
}

// ---------------------------------------------------------------------------
// Main: one workgroup per sequence, 16 wave32 per workgroup.
// y^T = alpha^T * Mt : A-operand = alpha chunk (lane=M layout replicates it
// across rows via plain ds_load_b128, no splat), B-operand = Mt 32x16 tile
// (lane=n, elements=k in table order -> 16 strided f16 loads/lane). D[m,n] = y[n0+n] in every
// lane, so each lane writes exactly one output. Double-buffered alpha in LDS,
// one barrier per step.
// ---------------------------------------------------------------------------
__global__ __launch_bounds__(512) void fsa_forward(
    const _Float16* __restrict__ Mt,     // (S, Q, Q) f16, Mt[s][j][i]
    const float*    __restrict__ initexp,
    const float*    __restrict__ finexp,
    const int*      __restrict__ xs,     // (B, T)
    float*          __restrict__ out) {  // (B,)
  __shared__ int      xs_sh[TT];
  __shared__ _Float16 alpha_h[2][QS];   // f16 copies fed to WMMA A operand
  __shared__ float    alpha_f[2][QS];   // f32 master
  __shared__ float    red[512];

  const int tid  = threadIdx.x;
  const int lane = tid & 31;
  const int w    = tid >> 5;       // wave id 0..15
  const int b    = blockIdx.x;     // sequence id
  const int hig  = lane >> 4;      // lane-group
  const int n0   = w * 32;         // this wave's 32 output states

  if (tid < TT) xs_sh[tid] = xs[b * TT + tid];
  if (tid < QS) {
    const float v = initexp[tid];
    alpha_f[0][tid] = v;
    alpha_h[0][tid] = (_Float16)v;
  }
  __syncthreads();

  for (int t = 0; t < TT; ++t) {
    const int rb = t & 1;          // read buffer
    const int wb = rb ^ 1;         // write buffer
    const _Float16* __restrict__ M = Mt + (size_t)xs_sh[t] * (QS * QS);

    // Hint next step's matrix (global_prefetch_b8); block spans the full 512 KB matrix.
    if (t + 1 < TT) {
      const char* nm = (const char*)(Mt + (size_t)xs_sh[t + 1] * (QS * QS));
      __builtin_prefetch(nm + (size_t)tid * 512, 0, 2);
      __builtin_prefetch(nm + 262144 + (size_t)tid * 512, 0, 2);
    }

    v8f acc0 = {};                 // outputs [n0,    n0+16)
    v8f acc1 = {};                 // outputs [n0+16, n0+32)

    #pragma unroll 4
    for (int kc = 0; kc < 16; ++kc) {      // K = 512 in chunks of 32
      const int kb = kc * 32;

      // A operand (16x32): lane=M, halves=K; lane-group 0 needs K {0..7,16..23},
      // group 1 needs {8..15,24..31} -> two ds_load_b128, replicated over rows.
      const _Float16* ap = &alpha_h[rb][kb + hig * 8];
      const v8h alo = *(const v8h*)ap;
      const v8h ahi = *(const v8h*)(ap + 16);
      const v16h am = __builtin_shufflevector(alo, ahi,
          0,1,2,3,4,5,6,7,8,9,10,11,12,13,14,15);

      // B operand (32x16): lane = n = lane%16, elements = k in table order (matches A).
      const _Float16* bp = M + (size_t)kb * QS + n0 + (lane & 15);
      v16h bm0, bm1;
      #pragma unroll
      for (int s = 0; s < 16; ++s) {
        const int k = (s < 8) ? (hig * 8 + s) : (16 + hig * 8 + (s - 8));
        bm0[s] = bp[(size_t)k * QS];
        bm1[s] = bp[(size_t)k * QS + 16];
      }

      acc0 = __builtin_amdgcn_wmma_f32_16x16x32_f16(
          false, am, false, bm0, (short)0, acc0, false, false);
      acc1 = __builtin_amdgcn_wmma_f32_16x16x32_f16(
          false, am, false, bm1, (short)0, acc1, false, false);
    }

    // D[m,n] = y[n0+n] identical in all rows m: lane holds its result at
    // N = lane%16 in every VGPR -> lanes 0-15 write tile0, 16-31 write tile1.
    const float y = hig ? acc1[0] : acc0[0];
    const int oi = n0 + lane;
    alpha_f[wb][oi] = y;
    alpha_h[wb][oi] = (_Float16)y;
    __syncthreads();               // single barrier per step (double-buffered)
  }

  // TT even -> final alpha sits in buffer 0. log(alpha_T . exp(final)).
  red[tid] = alpha_f[0][tid] * finexp[tid];
  __syncthreads();
  for (int off = 256; off > 0; off >>= 1) {
    if (tid < off) red[tid] += red[tid + off];
    __syncthreads();
  }
  if (tid == 0) out[b] = __logf(red[0]);
}

// ---------------------------------------------------------------------------
extern "C" void kernel_launch(void* const* d_in, const int* in_sizes, int n_in,
                              void* d_out, int out_size, void* d_ws, size_t ws_size,
                              hipStream_t stream) {
  const float* A     = (const float*)d_in[0];   // (Q, S, Q) f32 log-weights
  const float* initv = (const float*)d_in[1];   // (Q,)
  const float* finv  = (const float*)d_in[2];   // (Q,)
  const int*   xs    = (const int*)d_in[3];     // (B, T)
  float*       out   = (float*)d_out;           // (B,)

  _Float16* Mt = (_Float16*)d_ws;
  const size_t aBytes = (size_t)SS * QS * QS * sizeof(_Float16);  // 32 MB
  float* initexp = (float*)((char*)d_ws + aBytes);
  float* finexp  = initexp + QS;

  fsa_prep_A<<<dim3(QS / 32, QS / 32, SS), 256, 0, stream>>>(A, Mt);
  fsa_prep_vec<<<1, 512, 0, stream>>>(initv, finv, initexp, finexp);
  fsa_forward<<<BB, 512, 0, stream>>>(Mt, initexp, finexp, xs, out);
}
